// MultiHeadAttention_10033043603545
// MI455X (gfx1250) — matrix operand order repaired and checked by reading, not compiled, not run on hardware
//
#include <hip/hip_runtime.h>

// Problem constants (reference: B=4, S=2048, D=1024, H=16, DK=64)
#define B_  4
#define S_  2048
#define D_  1024
#define H_  16
#define DK_ 64

typedef __attribute__((ext_vector_type(8)))  __bf16 bf16x8;
typedef __attribute__((ext_vector_type(16))) __bf16 bf16x16;
typedef __attribute__((ext_vector_type(8)))  float  v8f;
typedef __attribute__((ext_vector_type(4)))  unsigned int u32x4;
typedef __attribute__((ext_vector_type(8)))  int i32x8;
typedef __attribute__((ext_vector_type(4)))  int i32x4;

// ---------------------------------------------------------------------------
// Fragment helpers (layouts used here, wave32; contiguous k order, not the ISA table order):
//  A-frag 16x32 bf16: lane = {row = lane&15, half = lane>>4}; lane holds
//    K in [16*half, 16*half+16)  (one contiguous run, matching B)
//  B-frag 32x16 bf16: lane&15 = column N; lanes 0-15 hold K 0..15,
//    lanes 16-31 hold K 16..31  (one contiguous 16-element run per lane)
//  C/D 16x16 f32: VGPR r holds row (r + 8*half), col = lane&15
// ---------------------------------------------------------------------------
__device__ __forceinline__ bf16x16 pack2(bf16x8 lo, bf16x8 hi) {
  return __builtin_shufflevector(lo, hi, 0,1,2,3,4,5,6,7,8,9,10,11,12,13,14,15);
}
__device__ __forceinline__ bf16x16 ldB16(const __bf16* p) {
  return pack2(*(const bf16x8*)p, *(const bf16x8*)(p + 8));
}
__device__ __forceinline__ bf16x16 ldA16(const __bf16* row32, int half) {
  return pack2(*(const bf16x8*)(row32 + 16 * half),
               *(const bf16x8*)(row32 + 16 * half + 8));
}
__device__ __forceinline__ v8f wmma_bf16(bf16x16 a, bf16x16 b, v8f c) {
  return __builtin_amdgcn_wmma_f32_16x16x32_bf16(
      false, a, false, b, (short)0, c, false, false);
}
__device__ __forceinline__ float redmax16(float v) {
#pragma unroll
  for (int m = 1; m < 16; m <<= 1) v = fmaxf(v, __shfl_xor(v, m, 32));
  return v;
}
__device__ __forceinline__ float redsum16(float v) {
#pragma unroll
  for (int m = 1; m < 16; m <<= 1) v += __shfl_xor(v, m, 32);
  return v;
}

// ---------------------------------------------------------------------------
// Tensor Data Mover: 2D tile load Global -> LDS.
// D# packing per CDNA5 ISA 8.3/8.4: group0 = {count=1, lds_addr,
// global_addr[56:0], type=2}; group1 = {data_size=2B, tensor dims, tile dims,
// dim0 stride}.  lds_addr = low 32 bits of the flat shared address (the
// flat->LDS aperture mapping keeps the LDS byte offset in addr[31:0]).
// Arity differs across toolchains (bridge doc): clang-22 = 5 args,
// clang-23 = 6 args.
// ---------------------------------------------------------------------------
__device__ __forceinline__ void tdm_load_2d(unsigned lds_off, const void* gaddr,
                                            unsigned tile_d0, unsigned tile_d1,
                                            unsigned tensor_d0, unsigned tensor_d1,
                                            unsigned stride0_elems) {
  const unsigned long long ga = (unsigned long long)(size_t)gaddr;
  u32x4 g0;
  g0[0] = 1u;                                               // count=1 (user D#)
  g0[1] = lds_off;                                          // lds_addr (bytes)
  g0[2] = (unsigned)ga;                                     // global_addr[31:0]
  g0[3] = (unsigned)((ga >> 32) & 0x01FFFFFFu) | (2u << 30);// [56:32] + type=2
  i32x8 g1;
  g1[0] = (int)(1u << 16);                                  // data_size=1 -> 2B
  g1[1] = (int)((tensor_d0 & 0xFFFFu) << 16);               // tensor_dim0 lo16
  g1[2] = (int)((tensor_d0 >> 16) | ((tensor_d1 & 0xFFFFu) << 16));
  g1[3] = (int)((tensor_d1 >> 16) | (tile_d0 << 16));       // tile_dim0
  g1[4] = (int)tile_d1;                                     // tile_dim1 (dim2=0)
  g1[5] = (int)stride0_elems;                               // dim0 stride lo32
  g1[6] = 0;
  g1[7] = 0;
  const i32x4 z4 = {0, 0, 0, 0};
#if __clang_major__ >= 23
  const i32x8 z8 = {0, 0, 0, 0, 0, 0, 0, 0};
  __builtin_amdgcn_tensor_load_to_lds(g0, g1, z4, z4, z8, 0);
#else
  __builtin_amdgcn_tensor_load_to_lds(g0, g1, z4, z4, 0);
#endif
}

// ---------------------------------------------------------------------------
// f32 -> bf16 conversion (grid-stride)
// ---------------------------------------------------------------------------
__global__ void cvt_f32_bf16(const float* __restrict__ src,
                             unsigned short* __restrict__ dst, int n) {
  __bf16* d = (__bf16*)dst;
  int i = blockIdx.x * blockDim.x + threadIdx.x;
  int stride = gridDim.x * blockDim.x;
  for (; i < n; i += stride) d[i] = (__bf16)src[i];
}

// ---------------------------------------------------------------------------
// Y = X @ W^T + bias.  X:[M=B*S, D] bf16, W:[D, D] bf16 row-major.
// Block: 256 threads = 8 waves; block tile 256(M) x 64(N); wave tile 32 x 64.
// The shared 64x32 W tile is staged into LDS once per k-step by the TDM
// (issued by wave 0), removing the 8-way duplicated B-fragment global loads.
// MODE 0: dst bf16 [B,H,S,DK]  (Q / K; scale folds 1/sqrt(DK) into Q)
// MODE 1: dst bf16 [B,H,DK,S]  (V transposed for the P x V B-fragment)
// MODE 2: dst f32  [M, D]      (final output projection)
// ---------------------------------------------------------------------------
template <int MODE>
__global__ __launch_bounds__(256) void gemm_bf16(
    const unsigned short* __restrict__ Xus, const unsigned short* __restrict__ Wus,
    const float* __restrict__ bias, void* __restrict__ dstv, float scale) {
  __shared__ __bf16 wtile[64 * 32];  // W rows n0..n0+63, K slice k0..k0+31

  const __bf16* X = (const __bf16*)Xus;
  const __bf16* W = (const __bf16*)Wus;

  const int lane   = threadIdx.x & 31;
  const int wave   = threadIdx.x >> 5;
  const int lane15 = lane & 15;
  const int half   = lane >> 4;
  const int m0 = blockIdx.x * 256 + wave * 32;
  const int n0 = blockIdx.y * 64;

  const __bf16* arow0 = X + (size_t)(m0 + lane15) * D_;
  const __bf16* arow1 = arow0 + (size_t)16 * D_;

  v8f acc[8] = {v8f{}, v8f{}, v8f{}, v8f{}, v8f{}, v8f{}, v8f{}, v8f{}};

  for (int k0 = 0; k0 < D_; k0 += 32) {
    __syncthreads();  // previous k-step finished reading wtile
    if (wave == 0) {
      tdm_load_2d((unsigned)(size_t)wtile, W + (size_t)n0 * D_ + k0,
                  /*tile_d0=*/32, /*tile_d1=*/64,
                  /*tensor_d0=*/D_, /*tensor_d1=*/D_, /*stride0=*/D_);
      __builtin_amdgcn_s_wait_tensorcnt(0);
    }
    __syncthreads();  // wtile populated

    __builtin_prefetch(arow0 + k0 + 256, 0, 3);  // global_prefetch_b8
    __builtin_prefetch(arow1 + k0 + 256, 0, 3);
    const bf16x16 a0 = ldA16(arow0 + k0, half);
    const bf16x16 a1 = ldA16(arow1 + k0, half);
#pragma unroll
    for (int j = 0; j < 4; ++j) {
      const bf16x16 b = ldB16(&wtile[(16 * j + lane15) * 32 + 16 * half]);
      acc[j]     = wmma_bf16(a0, b, acc[j]);
      acc[4 + j] = wmma_bf16(a1, b, acc[4 + j]);
    }
  }

#pragma unroll
  for (int mi = 0; mi < 2; ++mi) {
#pragma unroll
    for (int r = 0; r < 8; ++r) {
      const int m = m0 + 16 * mi + r + 8 * half;
      const int b = m >> 11;        // / S_
      const int s = m & (S_ - 1);
#pragma unroll
      for (int j = 0; j < 4; ++j) {
        const int n = n0 + 16 * j + lane15;
        const float v = (acc[4 * mi + j][r] + bias[n]) * scale;
        if (MODE == 0) {
          const int h = n >> 6, dk = n & (DK_ - 1);
          ((__bf16*)dstv)[(((size_t)b * H_ + h) * S_ + s) * DK_ + dk] = (__bf16)v;
        } else if (MODE == 1) {
          const int h = n >> 6, dk = n & (DK_ - 1);
          ((__bf16*)dstv)[(((size_t)b * H_ + h) * DK_ + dk) * S_ + s] = (__bf16)v;
        } else {
          ((float*)dstv)[(size_t)m * D_ + n] = v;
        }
      }
    }
  }
}

// ---------------------------------------------------------------------------
// Causal flash attention. One wave per 16-query tile; 8 waves per block walk
// consecutive q-tiles of the same (b,h) so K/V streams stay cache-resident.
// 64 keys per step: 8 score WMMAs (K-dim 64) + 8 P*V WMMAs; online softmax
// reductions amortized over 64 keys. P staged via LDS (C-layout -> A-frag).
// Q:[B,H,S,DK] (pre-scaled), K:[B,H,S,DK], Vt:[B,H,DK,S]; ctx bf16 [B,S,D].
// ---------------------------------------------------------------------------
__global__ __launch_bounds__(256) void flash_fwd(
    const unsigned short* __restrict__ Qus, const unsigned short* __restrict__ Kus,
    const unsigned short* __restrict__ Vus, unsigned short* __restrict__ ctxus) {
  __shared__ __bf16 pbuf[8][16 * 64];  // per-wave P tile (16 q-rows x 64 keys)

  const __bf16* Q  = (const __bf16*)Qus;
  const __bf16* K  = (const __bf16*)Kus;
  const __bf16* Vt = (const __bf16*)Vus;
  __bf16* ctx = (__bf16*)ctxus;

  const int lane   = threadIdx.x & 31;
  const int wave   = threadIdx.x >> 5;
  const int lane15 = lane & 15;
  const int half   = lane >> 4;

  const int tile = blockIdx.x * 8 + wave;  // 8192 q-tiles total
  const int qt = tile & (S_ / 16 - 1);
  const int bh = tile >> 7;                // (b*H + h)
  const int h  = bh & (H_ - 1);
  const int b  = bh >> 4;
  const int q0 = qt * 16;

  const __bf16* Qp = Q  + ((size_t)bh * S_ + q0) * DK_;
  const __bf16* Kp = K  + (size_t)bh * S_ * DK_;
  const __bf16* Vp = Vt + (size_t)bh * DK_ * S_;
  __bf16* pb = pbuf[wave];

  const __bf16* qrow = Qp + (size_t)lane15 * DK_;
  const bf16x16 a0 = ldA16(qrow,      half);  // dk 0..31
  const bf16x16 a1 = ldA16(qrow + 32, half);  // dk 32..63

  v8f o[4] = {v8f{}, v8f{}, v8f{}, v8f{}};
  float mi[8], li[8];
#pragma unroll
  for (int r = 0; r < 8; ++r) { mi[r] = -3.0e38f; li[r] = 0.0f; }

  const int ktEnd = (q0 + 79) >> 6;  // 64-key tiles for causal coverage
  for (int kt = 0; kt < ktEnd; ++kt) {
    const int kb = kt * 64;

    // scores: S = Q (16x64) x K^T (64x64), four 16x16 sub-tiles
    v8f s[4];
#pragma unroll
    for (int t = 0; t < 4; ++t) {
      const __bf16* kr = Kp + (size_t)(kb + 16 * t + lane15) * DK_;
      v8f st = {};
      st = wmma_bf16(a0, ldB16(kr + 16 * half),      st);
      st = wmma_bf16(a1, ldB16(kr + 32 + 16 * half), st);
      s[t] = st;
    }

    if (kb + 63 > q0) {  // diagonal block: elementwise causal mask
#pragma unroll
      for (int t = 0; t < 4; ++t)
#pragma unroll
        for (int r = 0; r < 8; ++r) {
          const int qr = q0 + r + 8 * half;
          s[t][r] = (kb + 16 * t + lane15 <= qr) ? s[t][r] : -1.0e9f;
        }
    }

    // online softmax update + stage P into LDS
    float cr[8];
#pragma unroll
    for (int r = 0; r < 8; ++r) {
      float mx = fmaxf(fmaxf(s[0][r], s[1][r]), fmaxf(s[2][r], s[3][r]));
      mx = redmax16(mx);
      const float mnew = fmaxf(mi[r], mx);
      const float p0 = __expf(s[0][r] - mnew);
      const float p1 = __expf(s[1][r] - mnew);
      const float p2 = __expf(s[2][r] - mnew);
      const float p3 = __expf(s[3][r] - mnew);
      const float c  = __expf(mi[r] - mnew);
      li[r] = li[r] * c + redsum16((p0 + p1) + (p2 + p3));
      mi[r] = mnew;
      cr[r] = c;
      __bf16* prow = pb + (r + 8 * half) * 64 + lane15;
      prow[0]  = (__bf16)p0;
      prow[16] = (__bf16)p1;
      prow[32] = (__bf16)p2;
      prow[48] = (__bf16)p3;
    }
#pragma unroll
    for (int r = 0; r < 8; ++r) {
      o[0][r] *= cr[r]; o[1][r] *= cr[r]; o[2][r] *= cr[r]; o[3][r] *= cr[r];
    }
    asm volatile("" ::: "memory");  // keep LDS stores before the re-load

    // P A-fragments (16x64 -> two 16x32 frags) from LDS
    const __bf16* prow = pb + lane15 * 64;
    const bf16x16 pa0 = ldA16(prow,      half);  // keys kb..kb+31
    const bf16x16 pa1 = ldA16(prow + 32, half);  // keys kb+32..kb+63

    // O += P (16x64) x V (64x64); Vt is [dk][s] so runs are contiguous
#pragma unroll
    for (int j = 0; j < 4; ++j) {
      const __bf16* vr = Vp + (size_t)(16 * j + lane15) * S_ + kb + 16 * half;
      o[j] = wmma_bf16(pa0, ldB16(vr),      o[j]);
      o[j] = wmma_bf16(pa1, ldB16(vr + 32), o[j]);
    }
  }

  // normalize and write ctx[b][s][h*64+dk]
#pragma unroll
  for (int r = 0; r < 8; ++r) {
    const float inv = 1.0f / li[r];
    const int sg = q0 + r + 8 * half;
    const size_t base = ((size_t)b * S_ + sg) * D_ + h * DK_ + lane15;
    ctx[base +  0] = (__bf16)(o[0][r] * inv);
    ctx[base + 16] = (__bf16)(o[1][r] * inv);
    ctx[base + 32] = (__bf16)(o[2][r] * inv);
    ctx[base + 48] = (__bf16)(o[3][r] * inv);
  }
}

// ---------------------------------------------------------------------------
// Launch: cvt(x, Wq, Wk, Wv, Wo) -> Q/K/Vt projections -> flash -> out proj
// Workspace map (bytes):
//   [ 0,16M) x bf16          [16M..24M) Wq/Wk/Wv/Wo bf16 (2MB each)
//   [24M,40M) Q [B,H,S,DK]   [40M,56M) K [B,H,S,DK]
//   [56M,72M) Vt [B,H,DK,S]  [72M,88M) ctx [B,S,D]
// ---------------------------------------------------------------------------
extern "C" void kernel_launch(void* const* d_in, const int* in_sizes, int n_in,
                              void* d_out, int out_size, void* d_ws, size_t ws_size,
                              hipStream_t stream) {
  const float* x  = (const float*)d_in[0];
  // d_in[1] = mask (causal mask computed analytically; not read)
  const float* Wq = (const float*)d_in[2];
  const float* bq = (const float*)d_in[3];
  const float* Wk = (const float*)d_in[4];
  const float* bk = (const float*)d_in[5];
  const float* Wv = (const float*)d_in[6];
  const float* bv = (const float*)d_in[7];
  const float* Wo = (const float*)d_in[8];
  const float* bo = (const float*)d_in[9];

  char* ws = (char*)d_ws;
  const size_t MB = 1024u * 1024u;
  unsigned short* xb  = (unsigned short*)(ws);
  unsigned short* wqb = (unsigned short*)(ws + 16 * MB);
  unsigned short* wkb = (unsigned short*)(ws + 18 * MB);
  unsigned short* wvb = (unsigned short*)(ws + 20 * MB);
  unsigned short* wob = (unsigned short*)(ws + 22 * MB);
  unsigned short* Qb  = (unsigned short*)(ws + 24 * MB);
  unsigned short* Kb  = (unsigned short*)(ws + 40 * MB);
  unsigned short* Vtb = (unsigned short*)(ws + 56 * MB);
  unsigned short* ctx = (unsigned short*)(ws + 72 * MB);

  const int nX = B_ * S_ * D_;  // 8M
  const int nW = D_ * D_;       // 1M
  cvt_f32_bf16<<<4096, 256, 0, stream>>>(x,  xb,  nX);
  cvt_f32_bf16<<<2048, 256, 0, stream>>>(Wq, wqb, nW);
  cvt_f32_bf16<<<2048, 256, 0, stream>>>(Wk, wkb, nW);
  cvt_f32_bf16<<<2048, 256, 0, stream>>>(Wv, wvb, nW);
  cvt_f32_bf16<<<2048, 256, 0, stream>>>(Wo, wob, nW);

  const float qscale = 0.125f;  // 1/sqrt(DK), folded into Q
  dim3 gp(B_ * S_ / 256, D_ / 64);  // (32, 16)
  gemm_bf16<0><<<gp, 256, 0, stream>>>(xb, wqb, bq, Qb,  qscale);
  gemm_bf16<0><<<gp, 256, 0, stream>>>(xb, wkb, bk, Kb,  1.0f);
  gemm_bf16<1><<<gp, 256, 0, stream>>>(xb, wvb, bv, Vtb, 1.0f);

  flash_fwd<<<(B_ * H_ * (S_ / 16)) / 8, 256, 0, stream>>>(Qb, Kb, Vtb, ctx);

  gemm_bf16<2><<<gp, 256, 0, stream>>>(ctx, wob, bo, d_out, 1.0f);
}
